// SwinTransformerBlock_54485955117812
// MI455X (gfx1250) — matrix operand order repaired and checked by reading, not compiled, not run on hardware
//
#include <hip/hip_runtime.h>
#include <hip/hip_bf16.h>
#include <math.h>

typedef __attribute__((ext_vector_type(16))) _Float16 v16h;
typedef __attribute__((ext_vector_type(8)))  _Float16 v8h;
typedef __attribute__((ext_vector_type(8)))  float    v8f;

namespace {

constexpr int kC   = 192;     // channels
constexpr int kNH  = 6;       // heads
constexpr int kHD  = 32;      // head dim
constexpr int kN   = 343;     // tokens per window (7^3)
constexpr int kNP  = 352;     // padded to 22*16
constexpr int kBW  = 256;     // B * windows = 4 * 64
constexpr int kM   = 87808;   // kBW * kN = B * L
constexpr int kL   = 21952;   // 28^3

__device__ __forceinline__ float wred(float v) {
  v += __shfl_xor(v, 1, 32);
  v += __shfl_xor(v, 2, 32);
  v += __shfl_xor(v, 4, 32);
  v += __shfl_xor(v, 8, 32);
  v += __shfl_xor(v, 16, 32);
  return v;
}

__device__ __forceinline__ int reg3(int c) { return (c < 21) ? 0 : ((c < 25) ? 1 : 2); }

// ---------- f32 -> f16 weight conversion ----------
__global__ __launch_bounds__(256) void cvt16_kernel(const float* __restrict__ src,
                                                    _Float16* __restrict__ dst, int n) {
  const int i = blockIdx.x * 256 + threadIdx.x;
  if (i < n) dst[i] = (_Float16)src[i];
}

// ---------- padded relative-position bias table: biasP[h][n(352)][m(352)] ----------
// pad entries (n>=343 or m>=343) get -1e30 so attention needs no bounds branches.
__global__ __launch_bounds__(256) void bias_table_kernel(const float* __restrict__ rel,
                                                         float* __restrict__ biasP) {
  const int idx = blockIdx.x * 256 + threadIdx.x;
  if (idx >= kNH * kNP * kNP) return;
  const int h = idx / (kNP * kNP);
  const int r = idx - h * kNP * kNP;
  const int n = r / kNP, m = r - (r / kNP) * kNP;
  float v = -1e30f;
  if (n < kN && m < kN) {
    const int zn = n / 49, yn = (n / 7) % 7, xn = n % 7;
    const int zm = m / 49, ym = (m / 7) % 7, xm = m % 7;
    const int ridx = ((zn - zm + 6) * 13 + (yn - ym + 6)) * 13 + (xn - xm + 6);
    v = rel[ridx * kNH + h];
  }
  biasP[idx] = v;
}

// ---------- LN1 + roll(-3,-3,-3) + window partition, f32 -> f16 ----------
__global__ __launch_bounds__(256) void ln1_window_kernel(
    const float* __restrict__ x, const float* __restrict__ g,
    const float* __restrict__ b, _Float16* __restrict__ xw16) {
  const int wave = threadIdx.x >> 5, lane = threadIdx.x & 31;
  const int gRow = blockIdx.x * 8 + wave;               // (b_, n) row
  const int bb   = gRow / kL;
  const int rem  = gRow - bb * kL;
  const int widx = rem / kN;
  const int n    = rem - widx * kN;
  const int wd = widx >> 4, wh = (widx >> 2) & 3, ww = widx & 3;
  const int z = n / 49, y = (n / 7) % 7, xx = n % 7;
  const int sd = (wd * 7 + z + 3) % 28;
  const int sh = (wh * 7 + y + 3) % 28;
  const int sw = (ww * 7 + xx + 3) % 28;
  const int l  = (sd * 28 + sh) * 28 + sw;
  const float* src = x + ((size_t)bb * kL + l) * kC;
  float v[6]; float s = 0.f;
#pragma unroll
  for (int i = 0; i < 6; ++i) { v[i] = src[lane + i * 32]; s += v[i]; }
  const float mean = wred(s) * (1.f / kC);
  float var = 0.f;
#pragma unroll
  for (int i = 0; i < 6; ++i) { float d = v[i] - mean; var += d * d; }
  var = wred(var) * (1.f / kC);
  const float inv = rsqrtf(var + 1e-5f);
  _Float16* dst = xw16 + (size_t)gRow * kC;
#pragma unroll
  for (int i = 0; i < 6; ++i) {
    const int c = lane + i * 32;
    dst[c] = (_Float16)((v[i] - mean) * inv * g[c] + b[c]);
  }
}

// ---------- LN2 (no permutation), f32 -> f16 ----------
__global__ __launch_bounds__(256) void ln2_kernel(
    const float* __restrict__ x2, const float* __restrict__ g,
    const float* __restrict__ b, _Float16* __restrict__ y16) {
  const int wave = threadIdx.x >> 5, lane = threadIdx.x & 31;
  const size_t gRow = (size_t)blockIdx.x * 8 + wave;
  const float* src = x2 + gRow * kC;
  float v[6]; float s = 0.f;
#pragma unroll
  for (int i = 0; i < 6; ++i) { v[i] = src[lane + i * 32]; s += v[i]; }
  const float mean = wred(s) * (1.f / kC);
  float var = 0.f;
#pragma unroll
  for (int i = 0; i < 6; ++i) { float d = v[i] - mean; var += d * d; }
  var = wred(var) * (1.f / kC);
  const float inv = rsqrtf(var + 1e-5f);
  _Float16* dst = y16 + gRow * kC;
#pragma unroll
  for (int i = 0; i < 6; ++i) {
    const int c = lane + i * 32;
    dst[c] = (_Float16)((v[i] - mean) * inv * g[c] + b[c]);
  }
}

// ---------- generic WMMA GEMM: (M x K) f16 @ (K x Nout) from f16 W (Nout,K) ----------
// MODE 0: QKV (+bias, scatter to [s][b_][h][n][d] f16)
// MODE 1: proj (+bias, window-reverse + roll(+3) + residual -> x2 f32)
// MODE 2: fc1 (+bias, exact GELU -> f16)
// MODE 3: fc2 (+bias, + residual x2 -> out f32)
struct GemmArgs {
  const _Float16* A;
  const _Float16* W16;   // (Nout, K) row-major f16
  const float* bias;
  const float* xres;
  float* xout;
  _Float16* out16;
  float* outf;
  int K;
};

template <int MODE>
__global__ __launch_bounds__(256) void wmma_gemm_kernel(GemmArgs p) {
  __shared__ _Float16 As[128 * 32];   // [row][k]
  __shared__ _Float16 Bs[64 * 32];    // [n][k]
  const int tid = threadIdx.x;
  const int wave = tid >> 5, lane = tid & 31;
  const int wm = wave & 3, wn = wave >> 2;          // 4 x 2 wave grid
  const int m0 = blockIdx.x * 128, n0 = blockIdx.y * 64;
  const int K = p.K;
  v8f acc[2][2];
#pragma unroll
  for (int i = 0; i < 2; ++i)
#pragma unroll
    for (int j = 0; j < 2; ++j)
#pragma unroll
      for (int r = 0; r < 8; ++r) acc[i][j][r] = 0.f;

  for (int k0 = 0; k0 < K; k0 += 32) {
    {  // A tile: 128 x 32 halves, 2 threads per row
      const int r = tid >> 1, cb = (tid & 1) * 16;
      const _Float16* ga = p.A + (size_t)(m0 + r) * K + k0 + cb;
      v8h* d = (v8h*)&As[r * 32 + cb];
      d[0] = *(const v8h*)ga;
      d[1] = *(const v8h*)(ga + 8);
      if (k0 + 32 < K) __builtin_prefetch(ga + 32, 0, 3);   // global_prefetch_b8
    }
    {  // B tile: Bs[n][kk] = W16[(n0+n)*K + k0+kk] (direct b128 copies)
      const int n = tid >> 2, kk0 = (tid & 3) * 8;
      const _Float16* gw = p.W16 + (size_t)(n0 + n) * K + k0 + kk0;
      *(v8h*)&Bs[n * 32 + kk0] = *(const v8h*)gw;
    }
    __syncthreads();

    const int kba = (lane >= 16) ? 8 : 0;   // A frag: lanes>=16 take K+8 blocks
    v16h af[2];
#pragma unroll
    for (int i = 0; i < 2; ++i) {
      const _Float16* s = &As[(wm * 32 + i * 16 + (lane & 15)) * 32];
      ((v8h*)&af[i])[0] = *(const v8h*)(s + kba);
      ((v8h*)&af[i])[1] = *(const v8h*)(s + kba + 16);
    }
    const int kbb = (lane >= 16) ? 8 : 0;   // B frag: lanes>=16 take K+8 blocks
    v16h bf[2];
#pragma unroll
    for (int j = 0; j < 2; ++j) {
      const _Float16* s = &Bs[(wn * 32 + j * 16 + (lane & 15)) * 32 + kbb];
      ((v8h*)&bf[j])[0] = *(const v8h*)s;
      ((v8h*)&bf[j])[1] = *(const v8h*)(s + 16);
    }
#pragma unroll
    for (int i = 0; i < 2; ++i)
#pragma unroll
      for (int j = 0; j < 2; ++j)
        acc[i][j] = __builtin_amdgcn_wmma_f32_16x16x32_f16(
            false, af[i], false, bf[j], (short)0, acc[i][j], false, false);
    __syncthreads();
  }

#pragma unroll
  for (int i = 0; i < 2; ++i)
#pragma unroll
    for (int j = 0; j < 2; ++j)
#pragma unroll
      for (int r = 0; r < 8; ++r) {
        const int gM = m0 + wm * 32 + i * 16 + r + ((lane >= 16) ? 8 : 0);
        const int gN = n0 + wn * 32 + j * 16 + (lane & 15);
        const float val = acc[i][j][r] + p.bias[gN];
        if constexpr (MODE == 0) {             // QKV scatter
          const int s3 = gN / 192, rm = gN - s3 * 192, h = rm >> 5, d = gN & 31;
          const int b_ = gM / 343, n = gM - b_ * 343;
          const size_t idx = ((((size_t)s3 * kBW + b_) * kNH + h) * kN + n) * kHD + d;
          p.out16[idx] = (_Float16)val;
        } else if constexpr (MODE == 1) {      // proj + reverse + roll + residual
          const int b_ = gM / 343, n = gM - b_ * 343;
          const int bb = b_ >> 6, widx = b_ & 63;
          const int wd = widx >> 4, wh = (widx >> 2) & 3, ww = widx & 3;
          const int z = n / 49, y = (n / 7) % 7, xx = n % 7;
          const int fd = (wd * 7 + z + 3) % 28;
          const int fh = (wh * 7 + y + 3) % 28;
          const int fw = (ww * 7 + xx + 3) % 28;
          const size_t idx = ((size_t)bb * kL + (fd * 28 + fh) * 28 + fw) * kC + gN;
          p.xout[idx] = p.xres[idx] + val;
        } else if constexpr (MODE == 2) {      // fc1 + exact GELU
          const float ge = 0.5f * val * (1.f + erff(val * 0.70710678118654752f));
          p.out16[(size_t)gM * 768 + gN] = (_Float16)ge;
        } else {                               // fc2 + residual
          const size_t idx = (size_t)gM * kC + gN;
          p.outf[idx] = p.xres[idx] + val;
        }
      }
}

// ---------- windowed attention: one block per (window*head, 64-row chunk) ----------
// LDS: Ksh[352][32] f16 | Vt[32][352] f16 | Psh[4][16][352] f16 | Ssh[4][16][352] f32
//      | invs[4][16] f32 | codes[352] int
__global__ __launch_bounds__(128) void attn_kernel(
    const _Float16* __restrict__ qkv, const float* __restrict__ biasP,
    _Float16* __restrict__ out16) {
  extern __shared__ __align__(16) char smem[];
  _Float16* Ksh = (_Float16*)smem;                       // 22528 B
  _Float16* Vt  = Ksh + kNP * kHD;                       // 22528 B
  _Float16* Psh = Vt + kHD * kNP;                        // 45056 B
  float*    Ssh = (float*)(Psh + 4 * 16 * kNP);          // 90112 B
  float*    invs = Ssh + 4 * 16 * kNP;                   // 256 B
  int*      codes = (int*)(invs + 4 * 16);               // 1408 B
  const int bh = blockIdx.x;                             // 0..1535
  const int b_ = bh / kNH, h = bh - b_ * kNH;
  const int tid = threadIdx.x, wave = tid >> 5, lane = tid & 31;
  const size_t sS = (size_t)kBW * kNH * kN * kHD;        // stride between q/k/v planes
  const _Float16* Qp = qkv + ((size_t)b_ * kNH + h) * kN * kHD;
  const _Float16* Kp = Qp + sS;
  const _Float16* Vp = Qp + 2 * sS;
  const int widx = b_ & 63;
  const int wd = widx >> 4, wh = (widx >> 2) & 3, ww = widx & 3;

  for (int idx = tid; idx < kNP * kHD; idx += 128) {
    const int r = idx >> 5, d = idx & 31;
    const bool ok = r < kN;
    Ksh[idx] = ok ? Kp[idx] : (_Float16)0.f;
    Vt[d * kNP + r] = ok ? Vp[idx] : (_Float16)0.f;      // V transposed: [d][k]
  }
  for (int n = tid; n < kNP; n += 128) {
    int code = 0;
    if (n < kN) {
      const int z = n / 49, y = (n / 7) % 7, xx = n % 7;
      code = reg3(wd * 7 + z) * 9 + reg3(wh * 7 + y) * 3 + reg3(ww * 7 + xx);
    }
    codes[n] = code;
  }
  __syncthreads();

  const int t = blockIdx.y * 4 + wave;             // 16-row query tile 0..21
  if (t >= 22) return;

  // Q fragment: A operand 16x32, K = full head dim
  const int kba = (lane >= 16) ? 8 : 0;
  const int hi8 = (lane >= 16) ? 8 : 0;
  const int qrow = t * 16 + (lane & 15);
  v16h qf;
#pragma unroll
  for (int u = 0; u < 16; ++u) qf[u] = (_Float16)0.f;
  if (qrow < kN) {
    const _Float16* s = Qp + (size_t)qrow * kHD;
    ((v8h*)&qf)[0] = *(const v8h*)(s + kba);
    ((v8h*)&qf)[1] = *(const v8h*)(s + kba + 16);
  }

  // per-row shift-mask codes for this wave's 8 accumulator rows
  int cn8[8];
#pragma unroll
  for (int r = 0; r < 8; ++r) cn8[r] = codes[t * 16 + r + hi8];

  float* Sw = Ssh + wave * 16 * kNP;
  _Float16* Pw = Psh + wave * 16 * kNP;
  const float scale = 0.17677669529663687f;        // 1/sqrt(32)
  const int kbb = (lane >= 16) ? 8 : 0;
  const float* bhp = biasP + ((size_t)h * kNP + t * 16) * kNP;  // padded, branch-free

  for (int kt = 0; kt < 22; ++kt) {
    const int m = kt * 16 + (lane & 15);
    const _Float16* s = &Ksh[m * kHD + kbb];
    v16h kf;
    ((v8h*)&kf)[0] = *(const v8h*)s;
    ((v8h*)&kf)[1] = *(const v8h*)(s + 16);
    v8f acc;
#pragma unroll
    for (int r = 0; r < 8; ++r) acc[r] = 0.f;
    acc = __builtin_amdgcn_wmma_f32_16x16x32_f16(false, qf, false, kf, (short)0,
                                                 acc, false, false);
    const int cm = codes[m];
#pragma unroll
    for (int r = 0; r < 8; ++r) {
      const int row = r + hi8;
      const float bias = bhp[row * kNP + m];       // -1e30 in padding rows/cols
      const float msk = (cn8[r] != cm) ? -100.f : 0.f;
      Sw[row * kNP + m] = fmaf(acc[r], scale, bias) + msk;
    }
  }
  asm volatile("s_wait_dscnt 0" ::: "memory");     // intra-wave LDS RAW before softmax

  {  // softmax: 2 lanes per row; store unnormalized exp as f16, defer 1/sum to output
    const int row = lane & 15;
    float mx = -1e30f;
    for (int c = (lane >> 4); c < kNP; c += 2) mx = fmaxf(mx, Sw[row * kNP + c]);
    mx = fmaxf(mx, __shfl_xor(mx, 16, 32));
    float sum = 0.f;
    for (int c = (lane >> 4); c < kNP; c += 2) {
      const float e = __expf(Sw[row * kNP + c] - mx);
      Pw[row * kNP + c] = (_Float16)e;
      sum += e;
    }
    sum += __shfl_xor(sum, 16, 32);
    if (lane < 16) invs[wave * 16 + row] = 1.f / sum;
  }
  asm volatile("s_wait_dscnt 0" ::: "memory");     // P/invs visible before reads

  // O = P @ V : K = 352 in 11 steps of 32; two 16-wide d tiles; b128 LDS loads only
  v8f o0, o1;
#pragma unroll
  for (int r = 0; r < 8; ++r) { o0[r] = 0.f; o1[r] = 0.f; }
  const int arow = lane & 15;
  for (int ksb = 0; ksb < 11; ++ksb) {
    const _Float16* ps = &Pw[arow * kNP + ksb * 32 + kba];
    v16h pf;
    ((v8h*)&pf)[0] = *(const v8h*)ps;
    ((v8h*)&pf)[1] = *(const v8h*)(ps + 16);
    const _Float16* v0p = &Vt[(lane & 15) * kNP + ksb * 32 + kbb];
    const _Float16* v1p = v0p + 16 * kNP;
    v16h vf0, vf1;
    ((v8h*)&vf0)[0] = *(const v8h*)v0p;
    ((v8h*)&vf0)[1] = *(const v8h*)(v0p + 16);
    ((v8h*)&vf1)[0] = *(const v8h*)v1p;
    ((v8h*)&vf1)[1] = *(const v8h*)(v1p + 16);
    o0 = __builtin_amdgcn_wmma_f32_16x16x32_f16(false, pf, false, vf0, (short)0, o0, false, false);
    o1 = __builtin_amdgcn_wmma_f32_16x16x32_f16(false, pf, false, vf1, (short)0, o1, false, false);
  }
#pragma unroll
  for (int r = 0; r < 8; ++r) {
    const int rowi = r + hi8;
    const int nq = t * 16 + rowi;
    if (nq < kN) {
      const float inv = invs[wave * 16 + rowi];
      _Float16* dst = out16 + ((size_t)b_ * kN + nq) * kC + h * kHD;
      dst[lane & 15] = (_Float16)(o0[r] * inv);
      dst[16 + (lane & 15)] = (_Float16)(o1[r] * inv);
    }
  }
}

}  // namespace

extern "C" void kernel_launch(void* const* d_in, const int* in_sizes, int n_in,
                              void* d_out, int out_size, void* d_ws, size_t ws_size,
                              hipStream_t stream) {
  (void)in_sizes; (void)n_in; (void)out_size; (void)ws_size;
  const float* x      = (const float*)d_in[0];
  const float* n1g    = (const float*)d_in[1];
  const float* n1b    = (const float*)d_in[2];
  const float* qkv_w  = (const float*)d_in[3];
  const float* qkv_b  = (const float*)d_in[4];
  const float* proj_w = (const float*)d_in[5];
  const float* proj_b = (const float*)d_in[6];
  const float* rel_t  = (const float*)d_in[7];
  const float* n2g    = (const float*)d_in[8];
  const float* n2b    = (const float*)d_in[9];
  const float* fc1_w  = (const float*)d_in[10];
  const float* fc1_b  = (const float*)d_in[11];
  const float* fc2_w  = (const float*)d_in[12];
  const float* fc2_b  = (const float*)d_in[13];
  float* out = (float*)d_out;

  // workspace layout (bytes), ~341 MB total:
  //   [0)          87808*192 f16 : xw16 -> attn_out16 -> y16 (reused)
  //   [33718272)   3*87808*192 f16 : qkv16
  //   [134873088)  87808*192 f32 : x2 (post-attention residual)
  //   [202309632)  87808*768 f16 : h16 (fc1 activations)
  //   [337182720)  6*352*352 f32 : padded bias table
  //   [340156416)  f16 weights: qkv | proj | fc1 | fc2
  char* ws = (char*)d_ws;
  _Float16* buf16a = (_Float16*)ws;
  _Float16* qkv16  = (_Float16*)(ws + (size_t)33718272);
  float*    x2     = (float*)(ws + (size_t)134873088);
  _Float16* h16    = (_Float16*)(ws + (size_t)202309632);
  float*    biasP  = (float*)(ws + (size_t)337182720);
  _Float16* qkvw16 = (_Float16*)(ws + (size_t)340156416);
  _Float16* projw16 = qkvw16 + 576 * 192;
  _Float16* fc1w16  = projw16 + 192 * 192;
  _Float16* fc2w16  = fc1w16 + 768 * 192;

  // 0) one-time-per-call precompute: f16 weights + padded bias table
  cvt16_kernel<<<(576 * 192 + 255) / 256, 256, 0, stream>>>(qkv_w, qkvw16, 576 * 192);
  cvt16_kernel<<<(192 * 192 + 255) / 256, 256, 0, stream>>>(proj_w, projw16, 192 * 192);
  cvt16_kernel<<<(768 * 192 + 255) / 256, 256, 0, stream>>>(fc1_w, fc1w16, 768 * 192);
  cvt16_kernel<<<(192 * 768 + 255) / 256, 256, 0, stream>>>(fc2_w, fc2w16, 192 * 768);
  bias_table_kernel<<<(kNH * kNP * kNP + 255) / 256, 256, 0, stream>>>(rel_t, biasP);

  // 1) LN1 + shift + window partition
  ln1_window_kernel<<<kM / 8, 256, 0, stream>>>(x, n1g, n1b, buf16a);

  // 2) QKV projection (87808 x 192) @ (192 x 576)
  GemmArgs a0{buf16a, qkvw16, qkv_b, nullptr, nullptr, qkv16, nullptr, 192};
  wmma_gemm_kernel<0><<<dim3(kM / 128, 9), 256, 0, stream>>>(a0);

  // 3) windowed attention (precomputed bias, LDS codes for mask, fused softmax)
  constexpr int attn_lds = 22528 + 22528 + 45056 + 90112 + 256 + 1408;  // 181888 B
  (void)hipFuncSetAttribute((const void*)attn_kernel,
                            hipFuncAttributeMaxDynamicSharedMemorySize, attn_lds);
  attn_kernel<<<dim3(kBW * kNH, 6), 128, attn_lds, stream>>>(qkv16, biasP, buf16a);

  // 4) output projection + window reverse + unshift + residual -> x2
  GemmArgs a1{buf16a, projw16, proj_b, x, x2, nullptr, nullptr, 192};
  wmma_gemm_kernel<1><<<dim3(kM / 128, 3), 256, 0, stream>>>(a1);

  // 5) LN2 -> f16
  ln2_kernel<<<kM / 8, 256, 0, stream>>>(x2, n2g, n2b, buf16a);

  // 6) fc1 + GELU
  GemmArgs a2{buf16a, fc1w16, fc1_b, nullptr, nullptr, h16, nullptr, 192};
  wmma_gemm_kernel<2><<<dim3(kM / 128, 12), 256, 0, stream>>>(a2);

  // 7) fc2 + residual -> out
  GemmArgs a3{h16, fc2w16, fc2_b, x2, nullptr, nullptr, out, 768};
  wmma_gemm_kernel<3><<<dim3(kM / 128, 3), 256, 0, stream>>>(a3);
}
